// model_encdec_24232205484429
// MI455X (gfx1250) — matrix operand order repaired and checked by reading, not compiled, not run on hardware
//
#include <hip/hip_runtime.h>
#include <hip/hip_bf16.h>

// ---------------------------------------------------------------------------
// Problem constants (from reference): B=512, L=256, D=512, CIN=3, COUT=16
// ---------------------------------------------------------------------------
#define BATCH   512
#define LSEQ    256
#define DHID    512
#define NGATES  2048        // 4*D, PyTorch gate order i,f,g,o
#define KPAD    544         // D (h) + 32 (x cols 512..527 + zero pad)
#define KTILES  17          // 544 / 32
#define NBLOCKS 32
#define THREADS 512         // 16 waves/block -> 512 waves total
#define MTPW    2           // M-tiles per wave (share B fragments)

typedef __attribute__((ext_vector_type(16))) __bf16 v16bf;
typedef __attribute__((ext_vector_type(8)))  __bf16 v8bf;
typedef __attribute__((ext_vector_type(8)))  float  v8f;

__device__ __forceinline__ v8f zero_v8f() {
    v8f z;
#pragma unroll
    for (int i = 0; i < 8; ++i) z[i] = 0.0f;
    return z;
}

// Branch-free tanh: native CDNA5 V_TANH_F32 when the builtin exists, else a
// divergence-free exp-based formulation (libm tanhf emits EXEC-mask branches).
__device__ __forceinline__ float fast_tanh(float x) {
#if __has_builtin(__builtin_amdgcn_tanhf)
    return __builtin_amdgcn_tanhf(x);
#else
    float ax = __builtin_fabsf(x);
    float e  = __expf(-2.0f * ax);
    float r  = (1.0f - e) * __builtin_amdgcn_rcpf(1.0f + e);
    return __builtin_copysignf(r, x);
#endif
}

// sigmoid(x) = 0.5*tanh(x/2) + 0.5 : one v_mul + one v_tanh_f32 + one v_fma.
// Avoids the IEEE-correct divide sequence (v_div_scale/v_div_fmas + Newton
// steps) that 1/(1+exp(-x)) lowered to in round 3.
__device__ __forceinline__ float sigm(float x) {
#if __has_builtin(__builtin_amdgcn_tanhf)
    return __builtin_fmaf(0.5f, __builtin_amdgcn_tanhf(0.5f * x), 0.5f);
#else
    return __builtin_amdgcn_rcpf(1.0f + __expf(-x));
#endif
}

// ---------------------------------------------------------------------------
// Preprocessing kernel 1: build K-major concatenated bf16 weights
//   Wt[k][n] (k<512)      = W_hh[n][k]
//   Wt[512+j][n] (j<in)   = W_ih[n][j]
//   rest                  = 0   (so garbage in padded A columns contributes 0)
// ---------------------------------------------------------------------------
__global__ void prep_weights(const float* __restrict__ enc_wih,
                             const float* __restrict__ enc_whh,
                             const float* __restrict__ dec_wih,
                             const float* __restrict__ dec_whh,
                             __bf16* __restrict__ Wenc,
                             __bf16* __restrict__ Wdec) {
    int tid = blockIdx.x * blockDim.x + threadIdx.x;
    if (tid >= KPAD * NGATES) return;
    int k = tid / NGATES;
    int n = tid - k * NGATES;
    float e = 0.0f, d = 0.0f;
    if (k < DHID) {
        e = enc_whh[n * DHID + k];
        d = dec_whh[n * DHID + k];
    } else {
        int kk = k - DHID;
        if (kk < 16) e = enc_wih[n * 16 + kk];
        if (kk < 3)  d = dec_wih[n * 3 + kk];
    }
    Wenc[tid] = (__bf16)e;
    Wdec[tid] = (__bf16)d;
}

// ---------------------------------------------------------------------------
// Preprocessing kernel 2: Conv1d(3->16,k=3,pad=1) + ReLU, store bf16 as
// xemb[t][b][c] so each timestep's [512 x 16] slab is contiguous.
// ---------------------------------------------------------------------------
__global__ void conv_embed(const float* __restrict__ traj,
                           const float* __restrict__ cw,
                           const float* __restrict__ cb,
                           __bf16* __restrict__ xemb) {
    int tid = blockIdx.x * blockDim.x + threadIdx.x;
    if (tid >= LSEQ * BATCH * 16) return;
    int c = tid & 15;
    int b = (tid >> 4) & (BATCH - 1);
    int t = tid >> 13;               // 16*512 = 8192 = 2^13
    float s = cb[c];
#pragma unroll
    for (int kk = 0; kk < 3; ++kk) {
        int l = t + kk - 1;
        if (l >= 0 && l < LSEQ) {
#pragma unroll
            for (int ci = 0; ci < 3; ++ci)
                s += traj[b * (LSEQ * 3) + l * 3 + ci] * cw[c * 9 + ci * 3 + kk];
        }
    }
    xemb[tid] = (__bf16)(s > 0.0f ? s : 0.0f);
}

// ---------------------------------------------------------------------------
// Preprocessing kernel 3: init A buffers ([h | x | pad], bf16) and barrier.
// A0 = [0 | x_0 | 0], A1 = 0.  Re-run every call => graph-replay safe.
// ---------------------------------------------------------------------------
__global__ void init_state(__bf16* __restrict__ A0, __bf16* __restrict__ A1,
                           const __bf16* __restrict__ xemb, int* __restrict__ bar) {
    int tid = blockIdx.x * blockDim.x + threadIdx.x;
    if (tid < BATCH * KPAD) {
        int b = tid / KPAD;
        int k = tid - b * KPAD;
        __bf16 v0 = (__bf16)0.0f;
        if (k >= DHID && k < DHID + 16) v0 = xemb[b * 16 + (k - DHID)];
        A0[tid] = v0;
        A1[tid] = (__bf16)0.0f;
    }
    if (tid < 2) bar[tid] = 0;
}

// ---------------------------------------------------------------------------
// Device-wide sense-reversal barrier (requires all NBLOCKS co-resident).
// ---------------------------------------------------------------------------
__device__ __forceinline__ void grid_sync(int* cnt, volatile int* gen) {
    __syncthreads();
    if (threadIdx.x == 0) {
        __threadfence();
        int g = *gen;
        if (atomicAdd(cnt, 1) == NBLOCKS - 1) {
            atomicExch(cnt, 0);
            __threadfence();
            atomicAdd((int*)gen, 1);
        } else {
            while (*gen == g) __builtin_amdgcn_s_sleep(2);
        }
        __threadfence();
    }
    __syncthreads();
}

// ---------------------------------------------------------------------------
// Per-step gates GEMM, two M-tiles per wave sharing the B fragments.
// acc[mt][g] += A(16x544, rows m0+16*mt..) x Wt(:, gate-g cols d0..d0+15)
//
// A fragment per documented CDNA5 16-bit A layout:
//   lane<16 : row m0+lane,    K = k0+{0..7} and k0+{16..23}
//   lane>=16: row m0+lane-16, K = k0+{8..15} and k0+{24..31}
// B fragment (K-major Wt[544][2048]): lane holds N = n0+(lane&15), same K set as its A lane.
//
// The laundered *integer offset* (always 0, but opaque to LLVM) makes the
// weight addresses loop-variant so LICM cannot hoist the read-only B-fragment
// loads out of the 512-step time loop (round 1: hoist + 2KB/lane scratch
// spill). Unlike laundering the pointer itself (round 2), the kernarg-derived
// pointer chain stays intact, so the loads keep the inferred GLOBAL address
// space (global_load_b128) instead of degrading to FLAT.
// ---------------------------------------------------------------------------
__device__ __forceinline__ void gemm_gates(const __bf16* __restrict__ Acur,
                                           const __bf16* __restrict__ Wt,
                                           int m0, int d0, int lane,
                                           v8f acc[MTPW][4]) {
    const int kboff = (lane >> 4) << 3;            // 0 or 8
    const __bf16* abase0 = Acur + (m0 + (lane & 15)) * KPAD + kboff;
#pragma unroll 1
    for (int kt = 0; kt < KTILES; ++kt) {
        const int k0 = kt * 32;

        union { v16bf v; v8bf h[2]; } ua[MTPW];
#pragma unroll
        for (int mt = 0; mt < MTPW; ++mt) {
            const __bf16* ap = abase0 + mt * 16 * KPAD + k0;
            ua[mt].h[0] = *(const v8bf*)(ap);
            ua[mt].h[1] = *(const v8bf*)(ap + 16);
        }

        int woff = 0;                                // launder: defeat LICM,
        asm volatile("" : "+s"(woff));               // keep global addrspace
        const __bf16* brow = Wt + ((size_t)(k0 + kboff) * NGATES + d0 + (lane & 15) + woff);
#pragma unroll
        for (int g = 0; g < 4; ++g) {
            v16bf bv;
#pragma unroll
            for (int s = 0; s < 16; ++s)
                bv[s] = brow[(size_t)((s & 7) + ((s >> 3) << 4)) * NGATES + g * DHID];
#pragma unroll
            for (int mt = 0; mt < MTPW; ++mt) {
                acc[mt][g] = __builtin_amdgcn_wmma_f32_16x16x32_bf16(
                    false, ua[mt].v, false, bv, (short)0, acc[mt][g], false, false);
            }
        }
    }
}

// ---------------------------------------------------------------------------
// LSTM pointwise update for one 16x16 tile; accumulator layout: element r of
// lane L is (m = m0 + r + 8*(L>>4), n = d0 + (L&15)). c stays in registers.
// ---------------------------------------------------------------------------
__device__ __forceinline__ void lstm_update(v8f acc[4], const float* __restrict__ bias,
                                            float creg[8], __bf16* __restrict__ Anext,
                                            float* __restrict__ h_out,
                                            float* __restrict__ c_out,
                                            int m0, int d0, int lane, bool dump_state) {
    const int nc = d0 + (lane & 15);
    const float bi  = bias[nc];
    const float bf_ = bias[DHID + nc];
    const float bg  = bias[2 * DHID + nc];
    const float bo  = bias[3 * DHID + nc];
    const int mofs = m0 + ((lane >> 4) << 3);
#pragma unroll
    for (int r = 0; r < 8; ++r) {
        const int m = mofs + r;
        float iv = sigm(acc[0][r] + bi);
        float fv = sigm(acc[1][r] + bf_);
        float gv = fast_tanh(acc[2][r] + bg);
        float ov = sigm(acc[3][r] + bo);
        float cn = fv * creg[r] + iv * gv;
        creg[r] = cn;
        float hn = ov * fast_tanh(cn);
        Anext[m * KPAD + nc] = (__bf16)hn;
        if (dump_state) {
            h_out[m * DHID + nc] = hn;
            c_out[m * DHID + nc] = cn;
        }
    }
}

// ---------------------------------------------------------------------------
// Persistent encoder+decoder kernel. 512 waves: wave w -> (m_group = w>>5
// covering rows 32*m_group..32*m_group+31, d_tile = w&31). Double-buffered A.
// ---------------------------------------------------------------------------
__global__ __launch_bounds__(THREADS, 1)
void lstm_encdec_persistent(const __bf16* __restrict__ xemb,
                            const __bf16* __restrict__ Wenc,
                            const __bf16* __restrict__ Wdec,
                            const float* __restrict__ enc_b,
                            const float* __restrict__ dec_b,
                            const float* __restrict__ fc_w,
                            const float* __restrict__ fc_b,
                            __bf16* __restrict__ A0,
                            __bf16* __restrict__ A1,
                            float* __restrict__ pred,
                            float* __restrict__ h_out,
                            float* __restrict__ c_out,
                            int* __restrict__ bar) {
    const int lane = threadIdx.x & 31;
    const int wid = (blockIdx.x * THREADS + threadIdx.x) >> 5;   // 0..511
    const int m_group = wid >> 5;          // 0..15 -> rows m0..m0+31
    const int d_tile  = wid & 31;          // 0..31 -> cols d0..d0+15 per gate
    const int m0 = m_group * 32;
    const int d0 = d_tile * 16;
    int* cnt = bar;
    volatile int* gen = bar + 1;

    float creg[MTPW][8];
#pragma unroll
    for (int mt = 0; mt < MTPW; ++mt)
#pragma unroll
        for (int r = 0; r < 8; ++r) creg[mt][r] = 0.0f;

    unsigned T = 0;

    // ---------------- encoder: 256 steps, 1 barrier each ----------------
    for (int t = 0; t < LSEQ; ++t, ++T) {
        const __bf16* Acur = (T & 1u) ? A1 : A0;
        __bf16* Anext = (T & 1u) ? A0 : A1;

        v8f acc[MTPW][4];
#pragma unroll
        for (int mt = 0; mt < MTPW; ++mt)
#pragma unroll
            for (int g = 0; g < 4; ++g) acc[mt][g] = zero_v8f();

        gemm_gates(Acur, Wenc, m0, d0, lane, acc);

        const bool last = (t == LSEQ - 1);
#pragma unroll
        for (int mt = 0; mt < MTPW; ++mt)
            lstm_update(acc[mt], enc_b, creg[mt], Anext, h_out, c_out,
                        m0 + 16 * mt, d0, lane, last);

        // d_tile==0 wave stages next step's x columns for its 32 rows (or
        // zeros the decoder's zero-padding input at the last step).
        if (d_tile == 0) {
            __bf16* dst = Anext + (size_t)(m0 + lane) * KPAD + DHID;
            if (!last) {
                const __bf16* src = xemb + ((size_t)(t + 1) * BATCH + (m0 + lane)) * 16;
                *(v16bf*)dst = *(const v16bf*)src;
            } else {
                v16bf z;
#pragma unroll
                for (int i = 0; i < 16; ++i) z[i] = (__bf16)0.0f;
                *(v16bf*)dst = z;
            }
        }
        grid_sync(cnt, gen);
    }

    // ---------------- decoder: 256 steps, 2 barriers each ----------------
    for (int s = 0; s < LSEQ; ++s, ++T) {
        const __bf16* Acur = (T & 1u) ? A1 : A0;
        __bf16* Anext = (T & 1u) ? A0 : A1;

        v8f acc[MTPW][4];
#pragma unroll
        for (int mt = 0; mt < MTPW; ++mt)
#pragma unroll
            for (int g = 0; g < 4; ++g) acc[mt][g] = zero_v8f();

        gemm_gates(Acur, Wdec, m0, d0, lane, acc);
#pragma unroll
        for (int mt = 0; mt < MTPW; ++mt)
            lstm_update(acc[mt], dec_b, creg[mt], Anext, h_out, c_out,
                        m0 + 16 * mt, d0, lane, false);

        grid_sync(cnt, gen);   // h_s fully published in Anext

        // fc: coords = h_s @ fc_w^T + fc_b -> prediction[:, s, :] and the
        // bf16 feedback columns of Anext. 96 dot-products per 32-row group.
        if (d_tile == 0) {
#pragma unroll 1
            for (int idx = lane; idx < 96; idx += 32) {
                const int row = idx / 3;
                const int j = idx - row * 3;
                const __bf16* hv = Anext + (size_t)(m0 + row) * KPAD;
                const float* wrow = fc_w + j * DHID;
                float sum = fc_b[j];
#pragma unroll 1
                for (int d = 0; d < DHID; d += 8) {
                    v8bf hvv = *(const v8bf*)(hv + d);
#pragma unroll
                    for (int i = 0; i < 8; ++i)
                        sum += (float)hvv[i] * wrow[d + i];
                }
                pred[((size_t)(m0 + row) * LSEQ + s) * 3 + j] = sum;
                Anext[(size_t)(m0 + row) * KPAD + DHID + j] = (__bf16)sum;
            }
        }
        grid_sync(cnt, gen);   // feedback columns visible before next GEMM
    }
}

// ---------------------------------------------------------------------------
// Launch
// ---------------------------------------------------------------------------
extern "C" void kernel_launch(void* const* d_in, const int* in_sizes, int n_in,
                              void* d_out, int out_size, void* d_ws, size_t ws_size,
                              hipStream_t stream) {
    (void)in_sizes; (void)n_in; (void)out_size; (void)ws_size;

    const float* traj    = (const float*)d_in[0];
    const float* conv_w  = (const float*)d_in[1];
    const float* conv_b  = (const float*)d_in[2];
    const float* enc_wih = (const float*)d_in[3];
    const float* enc_whh = (const float*)d_in[4];
    const float* enc_b   = (const float*)d_in[5];
    const float* dec_wih = (const float*)d_in[6];
    const float* dec_whh = (const float*)d_in[7];
    const float* dec_b   = (const float*)d_in[8];
    const float* fc_w    = (const float*)d_in[9];
    const float* fc_b    = (const float*)d_in[10];

    char* ws = (char*)d_ws;
    size_t off = 0;
    auto alloc = [&](size_t bytes) -> char* {
        char* p = ws + off;
        off = (off + bytes + 255) & ~(size_t)255;
        return p;
    };
    __bf16* Wenc = (__bf16*)alloc((size_t)KPAD * NGATES * sizeof(__bf16)); // ~2.2 MB
    __bf16* Wdec = (__bf16*)alloc((size_t)KPAD * NGATES * sizeof(__bf16)); // ~2.2 MB
    __bf16* A0   = (__bf16*)alloc((size_t)BATCH * KPAD * sizeof(__bf16));  // ~0.56 MB
    __bf16* A1   = (__bf16*)alloc((size_t)BATCH * KPAD * sizeof(__bf16));  // ~0.56 MB
    __bf16* xemb = (__bf16*)alloc((size_t)LSEQ * BATCH * 16 * sizeof(__bf16)); // 4 MB
    int*    bar  = (int*)alloc(256);

    float* pred  = (float*)d_out;                       // [512][256][3]
    float* h_out = pred + (size_t)BATCH * LSEQ * 3;     // [512][512]
    float* c_out = h_out + (size_t)BATCH * DHID;        // [512][512]

    {   // weights -> K-major bf16 (with folded input projection + zero pad)
        int n = KPAD * NGATES;
        prep_weights<<<(n + 255) / 256, 256, 0, stream>>>(enc_wih, enc_whh,
                                                          dec_wih, dec_whh,
                                                          Wenc, Wdec);
    }
    {   // conv front-end -> bf16 time-major embeddings
        int n = LSEQ * BATCH * 16;
        conv_embed<<<(n + 255) / 256, 256, 0, stream>>>(traj, conv_w, conv_b, xemb);
    }
    {   // A buffers + barrier init
        int n = BATCH * KPAD;
        init_state<<<(n + 255) / 256, 256, 0, stream>>>(A0, A1, xemb, bar);
    }
    lstm_encdec_persistent<<<NBLOCKS, THREADS, 0, stream>>>(
        xemb, Wenc, Wdec, enc_b, dec_b, fc_w, fc_b,
        A0, A1, pred, h_out, c_out, bar);
}
